// AttentionFusion_87170656240167
// MI455X (gfx1250) — matrix operand order repaired and checked by reading, not compiled, not run on hardware
//
#include <hip/hip_runtime.h>
#include <math.h>

// ---------------------------------------------------------------------------
// Types for CDNA5 WMMA (wave32). 16x16x32 bf16 -> f32 accumulate.
// ---------------------------------------------------------------------------
typedef __attribute__((ext_vector_type(16))) __bf16 v16bf_t;
typedef __attribute__((ext_vector_type(8)))  float  v8f_t;
typedef __attribute__((ext_vector_type(4)))  unsigned int u32x4_t;
typedef __attribute__((ext_vector_type(8)))  int i32x8_t;
typedef __attribute__((ext_vector_type(4)))  int i32x4_t;

#define B_N   16384
#define H_N   512
#define S_N   3
#define NH_N  4
#define HD_N  128
#define FF_N  2048
#define R_N   (B_N * S_N)   // 49152 rows of the 3-token sequence

__device__ __forceinline__ unsigned short f2b(float f) {
    unsigned u = __float_as_uint(f);
    u += 0x7FFFu + ((u >> 16) & 1u);            // round-to-nearest-even
    return (unsigned short)(u >> 16);
}
__device__ __forceinline__ void unpack4(uint2 u, float* f) {
    f[0] = __uint_as_float((u.x & 0xffffu) << 16);
    f[1] = __uint_as_float(u.x & 0xffff0000u);
    f[2] = __uint_as_float((u.y & 0xffffu) << 16);
    f[3] = __uint_as_float(u.y & 0xffff0000u);
}
__device__ __forceinline__ float wave_sum(float v) {
    for (int o = 16; o > 0; o >>= 1) v += __shfl_xor(v, o, 32);
    return v;
}

// gfx1250 async global->LDS copy (ASYNCcnt-tracked, no VGPR round trip).
__device__ __forceinline__ void async_ld_b128(unsigned lds_addr, const void* gaddr) {
    asm volatile("global_load_async_to_lds_b128 %0, %1, off"
                 :: "v"(lds_addr), "v"(gaddr) : "memory");
}
__device__ __forceinline__ void wait_async_le1() {
    asm volatile("s_wait_asynccnt 0x1" ::: "memory");
}
__device__ __forceinline__ void wait_async_0() {
    asm volatile("s_wait_asynccnt 0x0" ::: "memory");
}

// ---------------------------------------------------------------------------
// Tensor Data Mover: 2D tile load Global->LDS (TENSORcnt-tracked).
// Loads `rows` x 32 bf16 elements; LDS rows padded 16 DWORDs data + 8 DWORDs
// pad = 48-half stride (pad_interval=3 -> every 16 DW, pad_amount=7 -> +8 DW).
// Descriptor per CDNA5 ISA 8.3-8.6 (D# group0/group1; groups 2,3 unused=0).
// ---------------------------------------------------------------------------
__device__ __forceinline__ void tdm_load_tile32(unsigned lds_addr, const void* gaddr,
                                                int rows, int stride_elems, int tensor_rows) {
    unsigned long long ga = (unsigned long long)gaddr;
    u32x4_t g0;
    g0[0] = 1u;                                            // count=1, user descriptor
    g0[1] = lds_addr;                                      // lds_addr[31:0]
    g0[2] = (unsigned)(ga & 0xffffffffu);                  // global_addr[31:0]
    g0[3] = (unsigned)((ga >> 32) & 0x01ffffffu)           // global_addr[56:32]
          | (2u << 30);                                    // type = 2 ("image")
    i32x8_t g1;
    g1[0] = (int)((1u << 16)                               // data_size = 2 bytes
                | (1u << 20)                               // pad_enable
                | (3u << 22)                               // pad_interval: 16 DWORDs
                | (7u << 25));                             // pad_amount: 8 DWORDs
    g1[1] = (int)(((unsigned)stride_elems & 0xffffu) << 16);           // tensor_dim0 lo16
    g1[2] = (int)(((unsigned)stride_elems >> 16)
                | (((unsigned)tensor_rows & 0xffffu) << 16));          // dim0 hi / dim1 lo
    g1[3] = (int)(((unsigned)tensor_rows >> 16) | (32u << 16));        // dim1 hi / tile_dim0=32
    g1[4] = rows;                                          // tile_dim1 (tile_dim2 = 0)
    g1[5] = stride_elems;                                  // tensor_dim0_stride lo32
    g1[6] = 0;                                             // stride hi / dim1_stride lo
    g1[7] = 0;
    i32x4_t z4 = {0, 0, 0, 0};
#if defined(__clang_major__) && (__clang_major__ >= 23)
    i32x8_t z8 = {0, 0, 0, 0, 0, 0, 0, 0};
    __builtin_amdgcn_tensor_load_to_lds(g0, g1, z4, z4, z8, 0);
#else
    __builtin_amdgcn_tensor_load_to_lds(g0, g1, z4, z4, 0);
#endif
}

// ---------------------------------------------------------------------------
// f32 -> bf16 strided convert
// ---------------------------------------------------------------------------
__global__ __launch_bounds__(256)
void cvt_f32_bf16(const float* __restrict__ src, unsigned short* __restrict__ dst,
                  int rows, int cols, int lds_, int ldd_) {
    size_t i = (size_t)blockIdx.x * blockDim.x + threadIdx.x;
    size_t total = (size_t)rows * cols;
    if (i >= total) return;
    int r = (int)(i / cols), c = (int)(i % cols);
    dst[(size_t)r * ldd_ + c] = f2b(src[(size_t)r * lds_ + c]);
}

// ---------------------------------------------------------------------------
// WMMA bf16 GEMM:  C[M,N] = A[M,K] (row-major, lda) * W[N,K]^T + bias[N]
// 256 threads = 8 waves (2 x 4). Block tile 64(M) x 256(N), K-step 32.
// Per wave: 32x64 sub-tile = 2x4 WMMA per K-step.
// A tile staged via global_load_async_to_lds_b128 (1 per thread);
// B tile staged via the Tensor Data Mover (wave 0 issues one descriptor).
// Double-buffered LDS; fragment layouts per CDNA5 ISA 7.12.2.
// ---------------------------------------------------------------------------
#define TSTR 48                      // LDS row stride in halves (16B-aligned rows)
#define AS_HALVES (64 * TSTR)        // 64 x 32 A tile
#define BS_HALVES (256 * TSTR)       // 256 x 32 B tile

union FragAB { v16bf_t v; uint4 q[2]; };
union FragC  { v8f_t v; float f[8]; };

template<bool WF32, bool WBF16, bool RELU, bool EXTRA>
__global__ __launch_bounds__(256)
void gemm_bf16(const unsigned short* __restrict__ A, int lda,
               const unsigned short* __restrict__ Bw,   // [N,K] row-major
               const float* __restrict__ bias,
               float* __restrict__ Cf, unsigned short* __restrict__ Cb, int ldc,
               const float* __restrict__ extra, int lde,
               int M, int N, int K) {
    __shared__ unsigned short As[2][AS_HALVES];
    __shared__ unsigned short Bs[2][BS_HALVES];

    const int t    = threadIdx.x;
    const int wave = t >> 5;
    const int lane = t & 31;
    const int wm   = wave >> 2;      // 0..1  (M sub-block)
    const int wn   = wave & 3;       // 0..3  (N sub-block)
    const int m0   = blockIdx.x * 64;
    const int n0   = blockIdx.y * 256;

    const int ar = t >> 2;           // A tile row 0..63
    const int ac = (t & 3) * 8;      // A tile col chunk (8 halves)

    const int mrow = lane & 15;
    const int hi   = lane >> 4;

    const unsigned asb[2] = { (unsigned)(unsigned long long)(void*)&As[0][0],
                              (unsigned)(unsigned long long)(void*)&As[1][0] };
    const unsigned bsb[2] = { (unsigned)(unsigned long long)(void*)&Bs[0][0],
                              (unsigned)(unsigned long long)(void*)&Bs[1][0] };

    FragC acc[2][4];
    #pragma unroll
    for (int i = 0; i < 2; ++i)
        #pragma unroll
        for (int j = 0; j < 4; ++j)
            #pragma unroll
            for (int r = 0; r < 8; ++r) acc[i][j].f[r] = 0.0f;

    auto issue_tile = [&](int k0, int buf) {
        // A: one async b128 per thread
        const unsigned short* gA = A + (size_t)(m0 + ar) * lda + k0 + ac;
        async_ld_b128(asb[buf] + (unsigned)(ar * TSTR + ac) * 2, gA);
        // B: one TDM descriptor per block (wave 0), 256x32 tile, padded rows
        if (wave == 0) {
            tdm_load_tile32(bsb[buf], Bw + (size_t)n0 * K + k0, 256, K, N);
        }
    };

    issue_tile(0, 0);

    const int nsteps = K >> 5;
    for (int s = 0; s < nsteps; ++s) {
        const int p = s & 1;
        if (s + 1 < nsteps) {
            issue_tile((s + 1) << 5, p ^ 1);
            wait_async_le1();                         // tile p's A landed
            __builtin_amdgcn_s_wait_tensorcnt(1);     // tile p's B landed
        } else {
            wait_async_0();
            __builtin_amdgcn_s_wait_tensorcnt(0);
        }
        __syncthreads();     // all waves see tile p in LDS

        const unsigned short* Ap = &As[p][0];
        const unsigned short* Bp = &Bs[p][0];
        FragAB fa[2], fb[4];
        #pragma unroll
        for (int i = 0; i < 2; ++i) {
            const unsigned short* r0 = &Ap[(wm * 32 + i * 16 + mrow) * TSTR + hi * 8];
            fa[i].q[0] = *reinterpret_cast<const uint4*>(r0);
            fa[i].q[1] = *reinterpret_cast<const uint4*>(r0 + 16);
        }
        #pragma unroll
        for (int j = 0; j < 4; ++j) {
            const unsigned short* r0 = &Bp[(wn * 64 + j * 16 + mrow) * TSTR + hi * 8];
            fb[j].q[0] = *reinterpret_cast<const uint4*>(r0);
            fb[j].q[1] = *reinterpret_cast<const uint4*>(r0 + 16);
        }
        #pragma unroll
        for (int i = 0; i < 2; ++i)
            #pragma unroll
            for (int j = 0; j < 4; ++j)
                acc[i][j].v = __builtin_amdgcn_wmma_f32_16x16x32_bf16(
                    false, fa[i].v, false, fb[j].v, (short)0, acc[i][j].v, false, false);
        __syncthreads();     // tile p fully consumed before its buffer is refilled
    }

    // Epilogue: VGPR r -> M = r + (hi?8:0); N = lane%16 within each 16x16 tile.
    #pragma unroll
    for (int i = 0; i < 2; ++i) {
        const int mbase = m0 + wm * 32 + i * 16 + hi * 8;
        #pragma unroll
        for (int j = 0; j < 4; ++j) {
            const int ncol = n0 + wn * 64 + j * 16 + mrow;
            const float bb = bias[ncol];
            #pragma unroll
            for (int r = 0; r < 8; ++r) {
                const int m = mbase + r;
                float val = acc[i][j].f[r] + bb;
                if (EXTRA) val += extra[(size_t)m * lde + ncol];
                if (RELU)  val = fmaxf(val, 0.0f);
                const size_t o = (size_t)m * ldc + ncol;
                if (WF32)  Cf[o] = val;
                if (WBF16) Cb[o] = f2b(val);
            }
        }
    }
}

// ---------------------------------------------------------------------------
// Attention over 3 tokens, one wave per (batch, head). qkv bf16 [R, 3H].
// ---------------------------------------------------------------------------
__global__ __launch_bounds__(256)
void attn3_kernel(const unsigned short* __restrict__ qkv, unsigned short* __restrict__ ctx) {
    const int gid  = blockIdx.x * 8 + (threadIdx.x >> 5);
    const int lane = threadIdx.x & 31;
    const int b = gid >> 2;
    const int h = gid & 3;
    const unsigned short* base = qkv + (size_t)b * 3 * (3 * H_N) + h * HD_N + lane * 4;

    float q[3][4], k[3][4], v[3][4];
    #pragma unroll
    for (int i = 0; i < 3; ++i) {
        const unsigned short* row = base + i * (3 * H_N);
        unpack4(*reinterpret_cast<const uint2*>(row),            q[i]);
        unpack4(*reinterpret_cast<const uint2*>(row + H_N),      k[i]);
        unpack4(*reinterpret_cast<const uint2*>(row + 2 * H_N),  v[i]);
    }
    float s[3][3];
    #pragma unroll
    for (int i = 0; i < 3; ++i)
        #pragma unroll
        for (int j = 0; j < 3; ++j) {
            float p = 0.f;
            #pragma unroll
            for (int t = 0; t < 4; ++t) p += q[i][t] * k[j][t];
            s[i][j] = wave_sum(p) * 0.08838834764831845f;  // 1/sqrt(128)
        }
    float aw[3][3];
    #pragma unroll
    for (int i = 0; i < 3; ++i) {
        float m = fmaxf(s[i][0], fmaxf(s[i][1], s[i][2]));
        float e0 = expf(s[i][0] - m), e1 = expf(s[i][1] - m), e2 = expf(s[i][2] - m);
        float inv = 1.0f / (e0 + e1 + e2);
        aw[i][0] = e0 * inv; aw[i][1] = e1 * inv; aw[i][2] = e2 * inv;
    }
    #pragma unroll
    for (int i = 0; i < 3; ++i) {
        unsigned short* orow = ctx + ((size_t)b * 3 + i) * H_N + h * HD_N + lane * 4;
        #pragma unroll
        for (int t = 0; t < 4; ++t) {
            float c = aw[i][0] * v[0][t] + aw[i][1] * v[1][t] + aw[i][2] * v[2][t];
            orow[t] = f2b(c);
        }
    }
}

// ---------------------------------------------------------------------------
// LayerNorm1: x = LN(comb + att) -> f32 (in-place over comb ok) + bf16 copy
// ---------------------------------------------------------------------------
__global__ __launch_bounds__(256)
void ln1_kernel(const float* __restrict__ comb, const float* __restrict__ att,
                const float* __restrict__ g, const float* __restrict__ be,
                float* __restrict__ xf, unsigned short* __restrict__ xb) {
    const int row  = blockIdx.x * 8 + (threadIdx.x >> 5);
    const int lane = threadIdx.x & 31;
    const size_t o = (size_t)row * H_N + lane * 16;
    float x[16], s = 0.f, s2 = 0.f;
    #pragma unroll
    for (int t = 0; t < 16; ++t) {
        x[t] = comb[o + t] + att[o + t];
        s += x[t]; s2 += x[t] * x[t];
    }
    s = wave_sum(s); s2 = wave_sum(s2);
    const float mu  = s * (1.0f / H_N);
    const float var = s2 * (1.0f / H_N) - mu * mu;
    const float r   = rsqrtf(var + 1e-5f);
    #pragma unroll
    for (int t = 0; t < 16; ++t) {
        const int c = lane * 16 + t;
        float y = (x[t] - mu) * r * g[c] + be[c];
        xf[o + t] = y;
        xb[o + t] = f2b(y);
    }
}

// ---------------------------------------------------------------------------
// LayerNorm2 + mean-pool over 3 tokens. One wave per batch element.
// ---------------------------------------------------------------------------
__global__ __launch_bounds__(256)
void ln2pool_kernel(const float* __restrict__ x1, const float* __restrict__ y2,
                    const float* __restrict__ g, const float* __restrict__ be,
                    float* __restrict__ pooled) {
    const int b    = blockIdx.x * 8 + (threadIdx.x >> 5);
    const int lane = threadIdx.x & 31;
    float accp[16];
    #pragma unroll
    for (int t = 0; t < 16; ++t) accp[t] = 0.f;
    for (int tok = 0; tok < 3; ++tok) {
        const size_t o = ((size_t)b * 3 + tok) * H_N + lane * 16;
        float x[16], s = 0.f, s2 = 0.f;
        #pragma unroll
        for (int t = 0; t < 16; ++t) {
            x[t] = x1[o + t] + y2[o + t];
            s += x[t]; s2 += x[t] * x[t];
        }
        s = wave_sum(s); s2 = wave_sum(s2);
        const float mu  = s * (1.0f / H_N);
        const float var = s2 * (1.0f / H_N) - mu * mu;
        const float r   = rsqrtf(var + 1e-5f);
        #pragma unroll
        for (int t = 0; t < 16; ++t) {
            const int c = lane * 16 + t;
            accp[t] += (x[t] - mu) * r * g[c] + be[c];
        }
    }
    #pragma unroll
    for (int t = 0; t < 16; ++t)
        pooled[(size_t)b * H_N + lane * 16 + t] = accp[t] * (1.0f / 3.0f);
}

// ---------------------------------------------------------------------------
// Host orchestration. Workspace requirement ~550 MB (with aliasing).
// ---------------------------------------------------------------------------
extern "C" void kernel_launch(void* const* d_in, const int* in_sizes, int n_in,
                              void* d_out, int out_size, void* d_ws, size_t ws_size,
                              hipStream_t stream) {
    (void)in_sizes; (void)n_in; (void)out_size; (void)ws_size;
    const float* visual  = (const float*)d_in[0];
    const float* tactile = (const float*)d_in[1];
    const float* arm     = (const float*)d_in[2];
    const float* Wv = (const float*)d_in[3];  const float* bv = (const float*)d_in[4];
    const float* Wt = (const float*)d_in[5];  const float* bt = (const float*)d_in[6];
    const float* Wa = (const float*)d_in[7];  const float* ba = (const float*)d_in[8];
    const float* in_proj_w = (const float*)d_in[9];
    const float* in_proj_b = (const float*)d_in[10];
    const float* out_w = (const float*)d_in[11]; const float* out_b = (const float*)d_in[12];
    const float* w1 = (const float*)d_in[13];    const float* b1 = (const float*)d_in[14];
    const float* w2 = (const float*)d_in[15];    const float* b2 = (const float*)d_in[16];
    const float* g1 = (const float*)d_in[17];    const float* be1 = (const float*)d_in[18];
    const float* g2 = (const float*)d_in[19];    const float* be2 = (const float*)d_in[20];
    const float* Wo = (const float*)d_in[21];    const float* bo = (const float*)d_in[22];
    float* out = (float*)d_out;

    unsigned char* ws = (unsigned char*)d_ws;
    size_t off = 0;
    auto alloc = [&](size_t bytes) -> size_t {
        size_t o = off;
        off += (bytes + 255) & ~(size_t)255;
        return o;
    };
    // bf16 weights
    unsigned short* wv_b  = (unsigned short*)(ws + alloc((size_t)H_N * H_N * 2));
    unsigned short* wt_b  = (unsigned short*)(ws + alloc((size_t)H_N * H_N * 2));
    unsigned short* wa_b  = (unsigned short*)(ws + alloc((size_t)H_N * H_N * 2));
    unsigned short* ipj_b = (unsigned short*)(ws + alloc((size_t)3 * H_N * H_N * 2));
    unsigned short* ow_b  = (unsigned short*)(ws + alloc((size_t)H_N * H_N * 2));
    unsigned short* w1_b  = (unsigned short*)(ws + alloc((size_t)FF_N * H_N * 2));
    unsigned short* w2_b  = (unsigned short*)(ws + alloc((size_t)H_N * FF_N * 2));
    unsigned short* wo_b  = (unsigned short*)(ws + alloc((size_t)H_N * 3 * H_N * 2));
    // bf16 activations
    unsigned short* xin_b = (unsigned short*)(ws + alloc((size_t)B_N * 3 * H_N * 2)); // [B,3H] concat
    unsigned short* qkv_b = (unsigned short*)(ws + alloc((size_t)R_N * 3 * H_N * 2));
    unsigned short* cc_b  = (unsigned short*)(ws + alloc((size_t)R_N * H_N * 2));     // comb_bf, then ctx
    unsigned short* ffn_b = qkv_b;  // alias: ffn activations overlay (dead) qkv+ctx regions exactly
    unsigned short* x1_b  = (unsigned short*)(ws + alloc((size_t)R_N * H_N * 2));
    // f32 activations
    float* comb_f = (float*)(ws + alloc((size_t)R_N * H_N * 4));
    float* x1_f   = comb_f;         // LN1 runs in-place per row
    float* att_f  = (float*)(ws + alloc((size_t)R_N * H_N * 4));
    float* y2_f   = att_f;          // attention-out dead before ffn2 writes
    float* pooled = (float*)(ws + alloc((size_t)B_N * H_N * 4));

    auto cvt = [&](const float* s, unsigned short* d, int rows, int cols, int ls, int ld) {
        size_t tot = (size_t)rows * cols;
        cvt_f32_bf16<<<dim3((unsigned)((tot + 255) / 256)), dim3(256), 0, stream>>>(s, d, rows, cols, ls, ld);
    };
    // weights -> bf16
    cvt(Wv, wv_b, H_N, H_N, H_N, H_N);
    cvt(Wt, wt_b, H_N, H_N, H_N, H_N);
    cvt(Wa, wa_b, H_N, H_N, H_N, H_N);
    cvt(in_proj_w, ipj_b, 3 * H_N, H_N, H_N, H_N);
    cvt(out_w, ow_b, H_N, H_N, H_N, H_N);
    cvt(w1, w1_b, FF_N, H_N, H_N, H_N);
    cvt(w2, w2_b, H_N, FF_N, FF_N, FF_N);
    cvt(Wo, wo_b, H_N, 3 * H_N, 3 * H_N, 3 * H_N);
    // inputs -> bf16 packed [B, 3H] (visual|tactile|arm) == orig concat
    cvt(visual,  xin_b + 0 * H_N, B_N, H_N, H_N, 3 * H_N);
    cvt(tactile, xin_b + 1 * H_N, B_N, H_N, H_N, 3 * H_N);
    cvt(arm,     xin_b + 2 * H_N, B_N, H_N, H_N, 3 * H_N);

    // 1) Modality projections -> combined [B,3,H] (f32 + bf16), ldc = 3H, tok offset
    {
        dim3 g(B_N / 64, H_N / 256);
        gemm_bf16<true, true, false, false><<<g, 256, 0, stream>>>(
            xin_b + 0 * H_N, 3 * H_N, wv_b, bv,
            comb_f + 0 * H_N, cc_b + 0 * H_N, 3 * H_N, nullptr, 0, B_N, H_N, H_N);
        gemm_bf16<true, true, false, false><<<g, 256, 0, stream>>>(
            xin_b + 1 * H_N, 3 * H_N, wt_b, bt,
            comb_f + 1 * H_N, cc_b + 1 * H_N, 3 * H_N, nullptr, 0, B_N, H_N, H_N);
        gemm_bf16<true, true, false, false><<<g, 256, 0, stream>>>(
            xin_b + 2 * H_N, 3 * H_N, wa_b, ba,
            comb_f + 2 * H_N, cc_b + 2 * H_N, 3 * H_N, nullptr, 0, B_N, H_N, H_N);
    }

    // 2) QKV projection [R,3H] bf16
    gemm_bf16<false, true, false, false><<<dim3(R_N / 64, (3 * H_N) / 256), 256, 0, stream>>>(
        cc_b, H_N, ipj_b, in_proj_b, nullptr, qkv_b, 3 * H_N, nullptr, 0, R_N, 3 * H_N, H_N);

    // 3) Attention (ctx overwrites cc_b, which is dead after QKV)
    attn3_kernel<<<dim3((B_N * NH_N) / 8), 256, 0, stream>>>(qkv_b, cc_b);

    // 4) Output projection -> att_f
    gemm_bf16<true, false, false, false><<<dim3(R_N / 64, H_N / 256), 256, 0, stream>>>(
        cc_b, H_N, ow_b, out_b, att_f, nullptr, H_N, nullptr, 0, R_N, H_N, H_N);

    // 5) LN1 (in-place over comb_f) + bf16 copy
    ln1_kernel<<<dim3(R_N / 8), 256, 0, stream>>>(comb_f, att_f, g1, be1, x1_f, x1_b);

    // 6) FFN up + ReLU -> ffn_b (bf16, overlays qkv region)
    gemm_bf16<false, true, true, false><<<dim3(R_N / 64, FF_N / 256), 256, 0, stream>>>(
        x1_b, H_N, w1_b, b1, nullptr, ffn_b, FF_N, nullptr, 0, R_N, FF_N, H_N);

    // 7) FFN down -> y2_f
    gemm_bf16<true, false, false, false><<<dim3(R_N / 64, H_N / 256), 256, 0, stream>>>(
        ffn_b, FF_N, w2_b, b2, y2_f, nullptr, H_N, nullptr, 0, R_N, H_N, FF_N);

    // 8) LN2 + mean pool -> pooled [B,H]
    ln2pool_kernel<<<dim3(B_N / 8), 256, 0, stream>>>(x1_f, y2_f, g2, be2, pooled);

    // 9) Final: out = orig @ Wo^T + bo + pooled
    gemm_bf16<true, false, false, true><<<dim3(B_N / 64, H_N / 256), 256, 0, stream>>>(
        xin_b, 3 * H_N, wo_b, bo, out, nullptr, H_N, pooled, H_N, B_N, H_N, 3 * H_N);
}
